// COAT_block_66365834658250
// MI455X (gfx1250) — matrix operand order repaired and checked by reading, not compiled, not run on hardware
//
#include <hip/hip_runtime.h>

// ---------------------------------------------------------------------------
// ASTGCN block for MI455X (gfx1250, wave32, WMMA).
// Heavy GEMMs use v_wmma_f32_16x16x32_bf16, f32 accumulation, 2x2 tiles/wave.
// All GEMM B operands are pre-transposed to column-major bf16 so both A and
// Bt staging are unconditional coalesced 16B loads (no divergent u16 path).
// Workspace requirement: ~258 MB (bump-allocated below).
// ---------------------------------------------------------------------------

#define B_   8
#define N_   2048
#define FIN_ 3
#define T_   12
#define K_   3
#define FC_  64
#define FT_  64
#define FTM_ 768    // FC*T
#define DH_  384    // GAT head dim

typedef __attribute__((ext_vector_type(16))) __bf16 bf16x16;
typedef __attribute__((ext_vector_type(8)))  float  f32x8;

__device__ __forceinline__ __bf16 f2bf(float f) {
  union { float f; unsigned u; } x; x.f = f;
  unsigned r = x.u + 0x7FFFu + ((x.u >> 16) & 1u);   // RNE
  unsigned short h = (unsigned short)(r >> 16);
  union { unsigned short s; __bf16 b; } y; y.s = h;
  return y.b;
}

__device__ __forceinline__ float postop_apply(float v, int postop) {
  if (postop == 1)      return fmaxf(v, 0.f);
  else if (postop == 2) return (v > 0.f) ? v : (__expf(v) - 1.f);  // elu
  return v;
}

// ---------------------------------------------------------------------------
// Batched bf16 WMMA GEMM:
//   C[M,N](f32,rm) = A[M,K](bf16,rm) * Bt[N,K](bf16, column-major B)
// block = 256 threads = 8 waves (4x2); block tile 128(M) x 64(N); K step 32.
// Each wave computes 32x32 via four v_wmma_f32_16x16x32_bf16.
// Requires M%128==0, N%64==0, K%32==0 (true for all call sites:
//   M in {2048,16384}, N in {384,768}, K in {768,2048}).
// Batch z: Aoff=(z/aDiv)*aStride, Boff=(z%bMod)*bStride, Coff=z*cStride.
// ---------------------------------------------------------------------------
__global__ __launch_bounds__(256)
void gemm_bf16_wmma(const __bf16* __restrict__ A, const __bf16* __restrict__ Bt,
                    float* __restrict__ C, int M, int N, int K,
                    int lda, int ldbt, int ldc,
                    long aStride, int aDiv, long bStride, int bMod, long cStride,
                    int postop)
{
  __shared__ __align__(16) __bf16 sA[128][32];   // 8 KB
  __shared__ __align__(16) __bf16 sB[64][32];    // 4 KB, rows = n, cols = k

  int z = blockIdx.z;
  A  += (long)(z / aDiv) * aStride;
  Bt += (long)(z % bMod) * bStride;
  C  += (long)z * cStride;

  const int m0   = blockIdx.x * 128;
  const int n0   = blockIdx.y * 64;
  const int tid  = threadIdx.x;
  const int wave = tid >> 5;
  const int lane = tid & 31;
  const int wm   = (wave & 3) * 32;
  const int wn   = (wave >> 2) * 32;
  const int half = lane >> 4;
  const int l16  = lane & 15;

  // staging: A 128x32 -> 256 thr x 16 elems; Bt 64x32 -> 256 thr x 8 elems
  const int arow = tid >> 1;           // 0..127
  const int acol = (tid & 1) * 16;     // 0 or 16
  const int brow = tid >> 2;           // 0..63
  const int bcol = (tid & 3) * 8;      // 0,8,16,24

  const f32x8 z8 = {0.f,0.f,0.f,0.f,0.f,0.f,0.f,0.f};
  f32x8 acc00 = z8, acc01 = z8, acc10 = z8, acc11 = z8;

  const __bf16* aBase = A  + (long)(m0 + arow) * lda  + acol;
  const __bf16* bBase = Bt + (long)(n0 + brow) * ldbt + bcol;

  for (int kk = 0; kk < K; kk += 32) {
    *(float4*)(&sA[arow][acol])     = *(const float4*)(aBase + kk);
    *(float4*)(&sA[arow][acol + 8]) = *(const float4*)(aBase + kk + 8);
    *(float4*)(&sB[brow][bcol])     = *(const float4*)(bBase + kk);
    __syncthreads();

    if (kk + 32 < K) {                 // prefetch next tiles (global_prefetch_b8)
      __builtin_prefetch(aBase + kk + 32, 0, 3);
      __builtin_prefetch(bBase + kk + 32, 0, 3);
    }

    // A fragment (ISA 7.12.2, 16-bit 16x32): lanes 0-15 hold K 0-7 & 16-23,
    // lanes 16-31 hold K 8-15 & 24-31 -> two contiguous 16B chunks per lane.
    union Fr { bf16x16 v; float4 f[2]; };
    Fr a0, a1, b0, b1;
    a0.f[0] = *(const float4*)(&sA[wm + l16][half * 8]);
    a0.f[1] = *(const float4*)(&sA[wm + l16][16 + half * 8]);
    a1.f[0] = *(const float4*)(&sA[wm + 16 + l16][half * 8]);
    a1.f[1] = *(const float4*)(&sA[wm + 16 + l16][16 + half * 8]);
    // B fragment: same K layout as A (lanes 0-15 K 0-7 & 16-23, lanes 16-31 K 8-15 & 24-31).
    b0.f[0] = *(const float4*)(&sB[wn + l16][half * 8]);
    b0.f[1] = *(const float4*)(&sB[wn + l16][16 + half * 8]);
    b1.f[0] = *(const float4*)(&sB[wn + 16 + l16][half * 8]);
    b1.f[1] = *(const float4*)(&sB[wn + 16 + l16][16 + half * 8]);

    acc00 = __builtin_amdgcn_wmma_f32_16x16x32_bf16(false, a0.v, false, b0.v,
                                                    (short)0, acc00, false, false);
    acc01 = __builtin_amdgcn_wmma_f32_16x16x32_bf16(false, a0.v, false, b1.v,
                                                    (short)0, acc01, false, false);
    acc10 = __builtin_amdgcn_wmma_f32_16x16x32_bf16(false, a1.v, false, b0.v,
                                                    (short)0, acc10, false, false);
    acc11 = __builtin_amdgcn_wmma_f32_16x16x32_bf16(false, a1.v, false, b1.v,
                                                    (short)0, acc11, false, false);
    __syncthreads();
  }

  // C layout per tile: lane -> col = l16, VGPR i -> row = i + 8*half
  const int c0 = n0 + wn + l16;
  const int r0 = m0 + wm + half * 8;
  #pragma unroll
  for (int i = 0; i < 8; ++i) {
    C[(long)(r0 + i) * ldc + c0]            = postop_apply(acc00[i], postop);
    C[(long)(r0 + i) * ldc + c0 + 16]       = postop_apply(acc01[i], postop);
    C[(long)(r0 + 16 + i) * ldc + c0]       = postop_apply(acc10[i], postop);
    C[(long)(r0 + 16 + i) * ldc + c0 + 16]  = postop_apply(acc11[i], postop);
  }
}

// ---------------------------------------------------------------------------
// Generic batched transpose + f32->bf16 convert: src [z][R][C] -> dst [z][C][R]
// (used for cheb^T A-operand and for the W and Wh GEMM B-operands)
// ---------------------------------------------------------------------------
__global__ __launch_bounds__(256)
void transpose_cvt_bf16(const float* __restrict__ src, __bf16* __restrict__ dst,
                        int R, int C, long srcStride, long dstStride) {
  __shared__ float tile[32][33];
  int bz = blockIdx.z;
  const float* s = src + (long)bz * srcStride;
  __bf16*      d = dst + (long)bz * dstStride;
  int c0 = blockIdx.x * 32, r0 = blockIdx.y * 32;
  int tx = threadIdx.x, ty = threadIdx.y;            // 32 x 8
  for (int dy = 0; dy < 32; dy += 8) {
    int r = r0 + ty + dy;
    tile[ty + dy][tx] = (r < R && c0 + tx < C) ? s[(long)r * C + c0 + tx] : 0.f;
  }
  __syncthreads();
  for (int dy = 0; dy < 32; dy += 8) {
    int c = c0 + ty + dy;
    if (c < C && r0 + tx < R)
      d[(long)c * R + r0 + tx] = f2bf(tile[tx][ty + dy]);
  }
}

// x [B,N,36] f32 -> XpT [B*48, N] bf16 (column-major B for the cheb GEMM)
__global__ void pack_xT(const float* __restrict__ x, __bf16* __restrict__ XpT) {
  int idx = blockIdx.x * 256 + threadIdx.x;          // 384 * N_
  if (idx >= 384 * N_) return;
  int n  = idx % N_;
  int cf = idx / N_;                                 // 0..383 = b*48 + c
  int b  = cf / 48;
  int c  = cf % 48;
  float v = (c < 36) ? x[((long)b * N_ + n) * 36 + c] : 0.f;
  XpT[idx] = f2bf(v);
}

// lhs[k,i,o] = sum_j Gamma[j,i,o] * cos((j+0.5)*pi/K * k)
__global__ void make_lhs(const float* __restrict__ Gamma, float* __restrict__ lhs) {
  int idx = blockIdx.x * 256 + threadIdx.x;
  if (idx >= K_ * FIN_ * FC_) return;
  int o = idx % FC_;
  int i = (idx / FC_) % FIN_;
  int k = idx / (FC_ * FIN_);
  float s = 0.f;
  for (int j = 0; j < K_; ++j) {
    float ang = (j + 0.5f) * 3.14159265358979f / (float)K_ * (float)k;
    s += Gamma[(j * FIN_ + i) * FC_ + o] * __cosf(ang);
  }
  lhs[idx] = s;
}

// spatial_gcn = relu(0.5 * sum_{k,i} Y[k][m][b*48+i*T+t] * lhs[k,i,o])
__global__ void gcn_combine(const float* __restrict__ Y, const float* __restrict__ lhs,
                            float* __restrict__ h, __bf16* __restrict__ hbf) {
  long idx = blockIdx.x * 256L + threadIdx.x;        // B*N*FC*T
  if (idx >= (long)B_ * N_ * FC_ * T_) return;
  int  t  = (int)(idx % T_);
  long r  = idx / T_;
  int  o  = (int)(r % FC_);
  long bm = r / FC_;
  int  m  = (int)(bm % N_);
  int  b  = (int)(bm / N_);
  float s = 0.f;
  #pragma unroll
  for (int k = 0; k < K_; ++k)
    #pragma unroll
    for (int i = 0; i < FIN_; ++i)
      s += Y[((long)k * N_ + m) * 384 + b * 48 + i * T_ + t] *
           lhs[(k * FIN_ + i) * FC_ + o];
  s *= 0.5f;                      // 2/(K+1)
  s = s > 0.f ? s : 0.f;
  h[idx]   = s;
  hbf[idx] = f2bf(s);
}

__global__ void cvt_bf16(const float* __restrict__ src, __bf16* __restrict__ dst, long n) {
  long i = blockIdx.x * 256L + threadIdx.x;
  if (i < n) dst[i] = f2bf(src[i]);
}

// f1[r]=Wh[r,:].a[:D], f2[r]=Wh[r,:].a[D:], one wave per row
__global__ void att_fvec(const float* __restrict__ Wh, const float* __restrict__ a,
                         float* __restrict__ f1, float* __restrict__ f2,
                         int rows, int D) {
  int wave = (blockIdx.x * 256 + threadIdx.x) >> 5;
  int lane = threadIdx.x & 31;
  if (wave >= rows) return;
  const float* w = Wh + (long)wave * D;
  float s1 = 0.f, s2 = 0.f;
  for (int d = lane; d < D; d += 32) {
    float v = w[d];
    s1 += v * a[d];
    s2 += v * a[D + d];
  }
  #pragma unroll
  for (int o = 16; o > 0; o >>= 1) { s1 += __shfl_down(s1, o); s2 += __shfl_down(s2, o); }
  if (lane == 0) { f1[wave] = s1; f2[wave] = s2; }
}

// denom[b,j] = sum_i exp(sigmoid(lrelu(f1[b,i]+f2[b,j])))   (softmax axis=1)
__global__ void att_denom(const float* __restrict__ f1, const float* __restrict__ f2,
                          float* __restrict__ denom, float slope) {
  int b = blockIdx.y;
  int j = blockIdx.x * 256 + threadIdx.x;
  float fj = f2[b * N_ + j];
  const float* f1b = f1 + b * N_;
  float s = 0.f;
  for (int i = 0; i < N_; ++i) {
    float e = f1b[i] + fj;
    e = (e > 0.f) ? e : slope * e;
    float sg = 1.f / (1.f + __expf(-e));
    s += __expf(sg);
  }
  denom[b * N_ + j] = s;
}

// att[b,i,j] (bf16) = exp(sigmoid(lrelu(f1+f2))) / denom[b,j]
__global__ void att_fill(const float* __restrict__ f1, const float* __restrict__ f2,
                         const float* __restrict__ denom, __bf16* __restrict__ att,
                         float slope) {
  int b = blockIdx.z;
  int i = blockIdx.y;
  int j = blockIdx.x * 256 + threadIdx.x;
  float e = f1[b * N_ + i] + f2[b * N_ + j];
  e = (e > 0.f) ? e : slope * e;
  float sg = 1.f / (1.f + __expf(-e));
  float v  = __expf(sg) / denom[b * N_ + j];
  att[((long)b * N_ + i) * N_ + j] = f2bf(v);
}

// 1x3 temporal conv over T with zero pad, Cin=Cout=64, weights staged in LDS
__global__ __launch_bounds__(256)
void conv1x3(const float* __restrict__ X, const float* __restrict__ W,
             const float* __restrict__ bias, float* __restrict__ Yo) {
  __shared__ float sx[FC_ * T_];
  __shared__ float sw[FT_ * FC_ * 3];
  int row = blockIdx.x;                                // b*N + n
  const float* xr = X + (long)row * FC_ * T_;
  for (int i = threadIdx.x; i < FC_ * T_; i += 256) sx[i] = xr[i];
  for (int i = threadIdx.x; i < FT_ * FC_ * 3; i += 256) sw[i] = W[i];
  __syncthreads();
  for (int o = threadIdx.x; o < FT_ * T_; o += 256) {
    int f = o / T_, t = o % T_;
    float s = bias[f];
    const float* wf = &sw[f * FC_ * 3];
    #pragma unroll 4
    for (int c = 0; c < FC_; ++c) {
      float xm = (t > 0)      ? sx[c * T_ + t - 1] : 0.f;
      float x0 = sx[c * T_ + t];
      float xp = (t < T_ - 1) ? sx[c * T_ + t + 1] : 0.f;
      s += wf[c * 3] * xm + wf[c * 3 + 1] * x0 + wf[c * 3 + 2] * xp;
    }
    Yo[(long)row * FT_ * T_ + o] = s;
  }
}

__global__ void make_u1u2(const float* __restrict__ U1, const float* __restrict__ U2,
                          float* __restrict__ u1u2) {
  int f = threadIdx.x;
  if (f < FT_) {
    float s = 0.f;
    for (int g = 0; g < FT_; ++g) s += U1[f * FT_ + g] * U2[g];
    u1u2[f] = s;
  }
}

// lhsr[b,n,t] = sum_f tc*u1u2 ; rhst[b,n,t] = sum_f tc*U3
__global__ void tatt_vec(const float* __restrict__ tc, const float* __restrict__ u1u2,
                         const float* __restrict__ U3, float* __restrict__ lhsr,
                         float* __restrict__ rhst) {
  int idx = blockIdx.x * 256 + threadIdx.x;           // B*N*T
  if (idx >= B_ * N_ * T_) return;
  int row = idx / T_, t = idx % T_;
  const float* xr = tc + (long)row * FT_ * T_ + t;
  float s1 = 0.f, s2 = 0.f;
  for (int f = 0; f < FT_; ++f) {
    float v = xr[f * T_];
    s1 += v * u1u2[f];
    s2 += v * U3[f];
  }
  lhsr[idx] = s1;
  rhst[idx] = s2;
}

// prod[b,s,t] = sum_n lhsr[b,n,s]*rhst[b,n,t]
__global__ __launch_bounds__(256)
void temporal_prod(const float* __restrict__ lhsr, const float* __restrict__ rhst,
                   float* __restrict__ prod) {
  int id = blockIdx.x;                                 // b*144 + s*12 + t
  int b = id / 144, st = id % 144, s = st / T_, t = st % T_;
  const float* L = lhsr + (long)b * N_ * T_;
  const float* R = rhst + (long)b * N_ * T_;
  float acc = 0.f;
  for (int n = threadIdx.x; n < N_; n += 256)
    acc += L[n * T_ + s] * R[n * T_ + t];
  __shared__ float red[256];
  red[threadIdx.x] = acc; __syncthreads();
  for (int w = 128; w > 0; w >>= 1) {
    if (threadIdx.x < w) red[threadIdx.x] += red[threadIdx.x + w];
    __syncthreads();
  }
  if (threadIdx.x == 0) prod[id] = red[0];
}

// E[b,i,t] = softmax_i( sum_s Ve[i,s]*sigmoid(prod[b,s,t]+be[s,t]) )
__global__ void temporal_E(const float* __restrict__ prod, const float* __restrict__ be,
                           const float* __restrict__ Ve, float* __restrict__ E) {
  __shared__ float sg[144];
  __shared__ float e[144];
  int b = blockIdx.x, tid = threadIdx.x;
  if (tid < 144) {
    float v = prod[b * 144 + tid] + be[tid];
    sg[tid] = 1.f / (1.f + __expf(-v));
  }
  __syncthreads();
  if (tid < 144) {
    int i = tid / T_, t = tid % T_;
    float s = 0.f;
    for (int k = 0; k < T_; ++k) s += Ve[i * T_ + k] * sg[k * T_ + t];
    e[tid] = s;
  }
  __syncthreads();
  if (tid < T_) {
    int t = tid;
    float mx = -1e30f;
    for (int i = 0; i < T_; ++i) mx = fmaxf(mx, e[i * T_ + t]);
    float ssum = 0.f;
    for (int i = 0; i < T_; ++i) ssum += __expf(e[i * T_ + t] - mx);
    for (int i = 0; i < T_; ++i)
      E[b * 144 + i * T_ + t] = __expf(e[i * T_ + t] - mx) / ssum;
  }
}

// fused: tcn = tc @ E, + 1x1 residual conv, ReLU, LayerNorm over channels
__global__ __launch_bounds__(256)
void resid_ln(const float* __restrict__ tc, const float* __restrict__ E,
              const float* __restrict__ x, const float* __restrict__ res_w,
              const float* __restrict__ res_b, const float* __restrict__ ln_g,
              const float* __restrict__ ln_b, float* __restrict__ out) {
  __shared__ float stc[FT_ * T_];
  __shared__ float sE[144];
  __shared__ float sx[FIN_ * T_];
  __shared__ float zbuf[FT_ * T_];
  __shared__ float mu[T_], rstd[T_];
  int row = blockIdx.x;                                // b*N + n
  int b = row / N_;
  const float* tr = tc + (long)row * FT_ * T_;
  for (int i = threadIdx.x; i < FT_ * T_; i += 256) stc[i] = tr[i];
  for (int i = threadIdx.x; i < 144; i += 256) sE[i] = E[b * 144 + i];
  for (int i = threadIdx.x; i < FIN_ * T_; i += 256) sx[i] = x[(long)row * FIN_ * T_ + i];
  __syncthreads();
  for (int o = threadIdx.x; o < FT_ * T_; o += 256) {
    int f = o / T_, s = o % T_;
    float acc = res_b[f];
    #pragma unroll
    for (int i = 0; i < FIN_; ++i) acc += sx[i * T_ + s] * res_w[f * FIN_ + i];
    float tcn = 0.f;
    #pragma unroll
    for (int t = 0; t < T_; ++t) tcn += stc[f * T_ + t] * sE[t * T_ + s];
    float zv = acc + tcn;
    zbuf[o] = zv > 0.f ? zv : 0.f;
  }
  __syncthreads();
  if (threadIdx.x < T_) {
    int s = threadIdx.x;
    float m = 0.f;
    for (int f = 0; f < FT_; ++f) m += zbuf[f * T_ + s];
    m *= (1.f / FT_);
    float v = 0.f;
    for (int f = 0; f < FT_; ++f) { float d = zbuf[f * T_ + s] - m; v += d * d; }
    v *= (1.f / FT_);
    mu[s] = m; rstd[s] = rsqrtf(v + 1e-5f);
  }
  __syncthreads();
  for (int o = threadIdx.x; o < FT_ * T_; o += 256) {
    int f = o / T_, s = o % T_;
    out[(long)row * FT_ * T_ + o] = (zbuf[o] - mu[s]) * rstd[s] * ln_g[f] + ln_b[f];
  }
}

// ---------------------------------------------------------------------------
// Host orchestration
// ---------------------------------------------------------------------------
static inline void launch_gemm(const __bf16* A, const __bf16* Bt, float* C,
                               int M, int N, int K, int lda, int ldbt, int ldc,
                               long aStride, int aDiv, long bStride, int bMod,
                               long cStride, int batch, int postop, hipStream_t s) {
  dim3 grid(M / 128, N / 64, batch);
  gemm_bf16_wmma<<<grid, dim3(256), 0, s>>>(A, Bt, C, M, N, K, lda, ldbt, ldc,
                                            aStride, aDiv, bStride, bMod, cStride, postop);
}

static inline void launch_transpose(const float* src, __bf16* dst, int R, int C,
                                    long srcStride, long dstStride, int batch,
                                    hipStream_t s) {
  dim3 grid((C + 31) / 32, (R + 31) / 32, batch);
  transpose_cvt_bf16<<<grid, dim3(32, 8), 0, s>>>(src, dst, R, C, srcStride, dstStride);
}

extern "C" void kernel_launch(void* const* d_in, const int* in_sizes, int n_in,
                              void* d_out, int out_size, void* d_ws, size_t ws_size,
                              hipStream_t stream) {
  (void)in_sizes; (void)n_in; (void)out_size; (void)ws_size;
  const float* x       = (const float*)d_in[0];
  const float* cheb    = (const float*)d_in[1];
  const float* Gamma   = (const float*)d_in[2];
  const float* W_heads = (const float*)d_in[3];
  const float* a_heads = (const float*)d_in[4];
  const float* W_out   = (const float*)d_in[5];
  const float* a_out   = (const float*)d_in[6];
  const float* U1      = (const float*)d_in[7];
  const float* U2      = (const float*)d_in[8];
  const float* U3      = (const float*)d_in[9];
  const float* be      = (const float*)d_in[10];
  const float* Ve      = (const float*)d_in[11];
  const float* time_w  = (const float*)d_in[12];
  const float* time_b  = (const float*)d_in[13];
  const float* res_w   = (const float*)d_in[14];
  const float* res_b   = (const float*)d_in[15];
  const float* back_w  = (const float*)d_in[16];
  const float* back_b  = (const float*)d_in[17];
  const float* fore_w  = (const float*)d_in[18];
  const float* fore_b  = (const float*)d_in[19];
  const float* ln_g    = (const float*)d_in[20];
  const float* ln_b    = (const float*)d_in[21];
  float* out = (float*)d_out;

  // ---- workspace bump allocator (~258 MB total) ----
  char* base = (char*)d_ws;
  size_t off = 0;
  auto alloc = [&](size_t bytes) -> void* {
    off = (off + 255) & ~(size_t)255;
    void* p = base + off;
    off += bytes;
    return p;
  };
  __bf16* chebT  = (__bf16*)alloc(3L * N_ * N_ * 2);          // 25.2 MB (A operand)
  __bf16* XpT    = (__bf16*)alloc((size_t)N_ * 384 * 2);      //  1.6 MB (Bt: [384][2048])
  float*  Y      = (float*) alloc(3L * N_ * 384 * 4);         //  9.4 MB
  float*  lhs    = (float*) alloc(K_ * FIN_ * FC_ * 4);
  float*  bigA   = (float*) alloc(16384L * 768 * 4);          // 50.3 MB: h -> hcat -> gcn -> result
  __bf16* bigAbf = (__bf16*)alloc(16384L * 768 * 2);          // 25.2 MB: hbf -> hcatbf (A operands)
  float*  Whf    = (float*) alloc(16384L * 768 * 4);          // 50.3 MB: Wh / Whout -> tc
  __bf16* WhbfT  = (__bf16*)alloc(16384L * 768 * 2);          // 25.2 MB: [b][D][2048] (Bt)
  __bf16* wbfT   = (__bf16*)alloc(768L * 768 * 2);            //  1.2 MB: [N][K] (Bt)
  float*  f1v    = (float*) alloc(16384L * 4);
  float*  f2v    = (float*) alloc(16384L * 4);
  float*  denom  = (float*) alloc(16384L * 4);
  __bf16* att    = (__bf16*)alloc(8L * N_ * N_ * 2);          // 67.1 MB (A operand)
  float*  u1u2   = (float*) alloc(FT_ * 4);
  float*  lhsr   = (float*) alloc(16384L * T_ * 4);
  float*  rhst   = (float*) alloc(16384L * T_ * 4);
  float*  prodv  = (float*) alloc(8L * 144 * 4);
  float*  Ebuf   = (float*) alloc(8L * 144 * 4);
  float*  tc     = Whf;                                        // reuse (f32)
  float*  result = bigA;                                       // reuse (f32)

  // ---- stage A: ChebNetII spatial GCN ----
  // A operand = cheb^T (rhs contracts cheb over its FIRST spatial index)
  launch_transpose(cheb, chebT, N_, N_, (long)N_ * N_, (long)N_ * N_, 3, stream);
  pack_xT<<<(384 * N_ + 255) / 256, 256, 0, stream>>>(x, XpT);
  make_lhs<<<(K_ * FIN_ * FC_ + 255) / 256, 256, 0, stream>>>(Gamma, lhs);
  // Y[k] = chebT[k] @ Xp   (M=2048, N=384, K=2048), batch over k
  launch_gemm(chebT, XpT, Y, N_, 384, N_, N_, N_, 384,
              (long)N_ * N_, 1, 0, 1, (long)N_ * 384, 3, 0, stream);
  gcn_combine<<<(int)(((long)B_ * N_ * FC_ * T_ + 255) / 256), 256, 0, stream>>>(
      Y, lhs, bigA /*h*/, bigAbf /*hbf*/);

  // ---- stage B: GAT heads ----
  for (int hd = 0; hd < 2; ++hd) {
    // W [768,384] -> wbfT [384][768]
    launch_transpose(W_heads + (long)hd * 768 * DH_, wbfT, 768, DH_, 0, 0, 1, stream);
    // Wh = h @ W  (16384 x 384)
    launch_gemm(bigAbf, wbfT, Whf, 16384, DH_, 768, 768, 768, DH_,
                0, 1, 0, 1, 0, 1, 0, stream);
    // Wh [b*2048][384] -> WhbfT [b][384][2048]
    launch_transpose(Whf, WhbfT, N_, DH_, (long)N_ * DH_, (long)DH_ * N_, 8, stream);
    att_fvec<<<16384 / 8, 256, 0, stream>>>(Whf, a_heads + (long)hd * 2 * DH_,
                                            f1v, f2v, 16384, DH_);
    att_denom<<<dim3(N_ / 256, B_), 256, 0, stream>>>(f1v, f2v, denom, 0.1f);
    att_fill<<<dim3(N_ / 256, N_, B_), 256, 0, stream>>>(f1v, f2v, denom, att, 0.1f);
    // hcat[:, hd*384 : hd*384+384] = elu(att @ Wh), batched over b
    launch_gemm(att, WhbfT, bigA + hd * DH_, N_, DH_, N_, N_, N_, 768,
                (long)N_ * N_, 1, (long)DH_ * N_, 8, (long)N_ * 768, 8, 2, stream);
  }

  // ---- stage C: output attention ----
  cvt_bf16<<<(int)((16384L * 768 + 255) / 256), 256, 0, stream>>>(bigA, bigAbf, 16384L * 768);
  launch_transpose(W_out, wbfT, 768, 768, 0, 0, 1, stream);
  launch_gemm(bigAbf, wbfT, Whf, 16384, 768, 768, 768, 768, 768,
              0, 1, 0, 1, 0, 1, 0, stream);
  launch_transpose(Whf, WhbfT, N_, 768, (long)N_ * 768, 768L * N_, 8, stream);
  att_fvec<<<16384 / 8, 256, 0, stream>>>(Whf, a_out, f1v, f2v, 16384, 768);
  att_denom<<<dim3(N_ / 256, B_), 256, 0, stream>>>(f1v, f2v, denom, 0.1f);
  att_fill<<<dim3(N_ / 256, N_, B_), 256, 0, stream>>>(f1v, f2v, denom, att, 0.1f);
  // gcn_SAt = relu(att @ Wh_out)  -> bigA
  launch_gemm(att, WhbfT, bigA, N_, 768, N_, N_, N_, 768,
              (long)N_ * N_, 1, 768L * N_, 8, (long)N_ * 768, 8, 1, stream);

  // ---- stage D: temporal conv + temporal attention ----
  conv1x3<<<B_ * N_, 256, 0, stream>>>(bigA, time_w, time_b, tc);
  make_u1u2<<<1, 64, 0, stream>>>(U1, U2, u1u2);
  tatt_vec<<<(B_ * N_ * T_ + 255) / 256, 256, 0, stream>>>(tc, u1u2, U3, lhsr, rhst);
  temporal_prod<<<B_ * 144, 256, 0, stream>>>(lhsr, rhst, prodv);
  temporal_E<<<B_, 160, 0, stream>>>(prodv, be, Ve, Ebuf);

  // ---- stage E: residual + ReLU + LayerNorm (fused) ----
  resid_ln<<<B_ * N_, 256, 0, stream>>>(tc, Ebuf, x, res_w, res_b, ln_g, ln_b, result);

  // ---- stage F: backcast / forecast convs into d_out ----
  conv1x3<<<B_ * N_, 256, 0, stream>>>(result, back_w, back_b, out);
  conv1x3<<<B_ * N_, 256, 0, stream>>>(result, fore_w, fore_b,
                                       out + (long)B_ * N_ * FT_ * T_);
}
